// SpatialGraphConv_910533067547
// MI455X (gfx1250) — matrix operand order repaired and checked by reading, not compiled, not run on hardware
//
#include <hip/hip_runtime.h>
#include <stdint.h>

// ---------------- CDNA5 types ----------------
typedef __attribute__((ext_vector_type(16))) __bf16 v16bf;
typedef __attribute__((ext_vector_type(8)))  float  v8f;

#if __has_builtin(__builtin_amdgcn_global_load_async_to_lds_b128)
#define HAVE_ASYNC_LDS 1
#else
#define HAVE_ASYNC_LDS 0
#endif
#if __has_builtin(__builtin_amdgcn_s_wait_asynccnt)
#define HAVE_WAIT_ASYNC 1
#else
#define HAVE_WAIT_ASYNC 0
#endif

// async-copy builtin operand types (from hipcc diagnostic: int __vector(4) * with AS quals)
typedef int v4i __attribute__((vector_size(16)));
#if HAVE_ASYNC_LDS
typedef __attribute__((address_space(1))) v4i as1_v4i;
typedef __attribute__((address_space(3))) v4i as3_v4i;
#endif

// ---------------- problem dims ----------------
#define NB      64
#define CIN     64
#define COUT    64
#define TDIM    300
#define VDIM    25
#define KADJ    3
#define TVT     7500            // T*V
#define TTILE   16              // timesteps per block
#define STILE   400             // TTILE*VDIM (== 25 WMMA N-tiles exactly)
#define NELEM   30720000        // N*C*T*V
// ---------------- LDS layout (bytes / half-strides) ----------------
#define XS_STR  72              // halves per Xs row (400 rows of 64 ci)
#define WB_STR  72              // halves per W row  (192 rows of 64 ci)
#define YS_TSTR 40              // halves per (c,t) row (32 v, padded)
#define YS_CSTR 648             // halves per c (16*40+8, 16B aligned)
#define AB_STR  40              // halves per adjacency row (32 v)
#define STG_BYTES 102400        // f32 staging 64x400 (unioned with Ys)
#define YS_BYTES  82944         // 64*648*2
#define XS_OFF  102400
#define WB_OFF  160000          // XS_OFF + 400*72*2
#define AB_OFF  187648          // WB_OFF + 192*72*2
#define B2_OFF  195328          // AB_OFF + 96*40*2
#define LDS_TOTAL 203520        // B2_OFF + 64*32*4

// ---------------- WMMA fragment loaders (ISA 7.12.2 layouts) ----------------
union Frag { struct { uint4 lo; uint4 hi; } q; v16bf v; };

// A-operand, 16-bit 16x32: lane<16 holds K{0..7,16..23}, lane>=16 holds K{8..15,24..31}
__device__ __forceinline__ v16bf frag_a(const __bf16* row32, int lhalf) {
  Frag u;
  const char* p = (const char*)row32 + lhalf * 16;
  u.q.lo = *(const uint4*)(p);
  u.q.hi = *(const uint4*)(p + 32);
  return u.v;
}
// B-operand, 16-bit 32x16: lane<16 holds K{0..7,16..23}, lane>=16 holds K{8..15,24..31}
__device__ __forceinline__ v16bf frag_b(const __bf16* row32, int lhalf) {
  Frag u;
  const char* p = (const char*)row32 + lhalf * 16;
  u.q.lo = *(const uint4*)(p);
  u.q.hi = *(const uint4*)(p + 32);
  return u.v;
}

union BF2 { unsigned u; __bf16 h[2]; };
union BF4 { uint2 u; __bf16 h[4]; };

// =====================================================================
// K0: zero the stats accumulators
// =====================================================================
__global__ void k_zero_stats(float* stats) {
  if (threadIdx.x < 128) stats[threadIdx.x] = 0.0f;
}

// =====================================================================
// K1: fused (1x1 conv GEMM) -> (spatial aggregation GEMM) -> bf16 y
// grid (64 n, 19 t-tiles), 512 threads (16 wave32), ~200KB dynamic LDS
// =====================================================================
__global__ __launch_bounds__(512) void k_fused_gcn(
    const float* __restrict__ x, const float* __restrict__ Wg,
    const float* __restrict__ bg, const float* __restrict__ Aadj,
    const float* __restrict__ Eadj, unsigned short* __restrict__ ybf_raw)
{
  extern __shared__ char smem[];
  __bf16* ybf = (__bf16*)ybf_raw;

  const int tid  = threadIdx.x;
  const int lane = tid & 31;
  const int wid  = tid >> 5;
  const int l15  = lane & 15;
  const int lhalf = lane >> 4;
  const int n  = blockIdx.x;
  const int t0 = blockIdx.y * TTILE;

  float*  stg = (float*)(smem);           // staging f32 [64][400]
  __bf16* Ys  = (__bf16*)(smem);          // union: reused after conversion
  __bf16* Xs  = (__bf16*)(smem + XS_OFF); // [400 s][64 ci] (B-operand layout)
  __bf16* Wb  = (__bf16*)(smem + WB_OFF); // [192 o][64 ci] (A-operand layout)
  __bf16* Ab  = (__bf16*)(smem + AB_OFF); // [k][wtile][16 w][32 v]
  float*  B2  = (float*)(smem + B2_OFF);  // bias2[c][32]

  // ---- phase 0: stage x tile into LDS (async DMA path on gfx1250) ----
  {
    const char* xb = (const char*)x;
    const long long xmax = (long long)NELEM * 4 - 16;
    for (int it = 0; it < 13; ++it) {
      int ch = it * 512 + tid;                 // 6400 x 16B chunks
      if (ch < 6400) {
        int ci = ch / 100, part = ch % 100;
        long long gb = (((long long)n * CIN + ci) * TVT + (long long)t0 * VDIM) * 4
                       + (long long)part * 16;
        if (gb > xmax) gb = xmax;              // clamp tail (garbage t>=300 unused)
#if HAVE_ASYNC_LDS
        __builtin_amdgcn_global_load_async_to_lds_b128(
            (as1_v4i*)(xb + gb), (as3_v4i*)(smem + ch * 16), 0, 0);
#else
        *(uint4*)(smem + ch * 16) = *(const uint4*)(xb + gb);
#endif
      }
    }
#if HAVE_ASYNC_LDS
#if HAVE_WAIT_ASYNC
    __builtin_amdgcn_s_wait_asynccnt(0);
#else
    asm volatile("s_wait_asynccnt 0" ::: "memory");
#endif
#endif
  }
  __syncthreads();

  // ---- phase 1: build bf16 operands in LDS ----
  // Xs[s][ci] (transposed x tile)
  for (int it = 0; it < 25; ++it) {
    int idx = it * 512 + tid;                 // 12800 pairs
    int cip = (idx & 31) << 1;
    int s   = idx >> 5;
    BF2 p;
    p.h[0] = (__bf16)stg[cip * STILE + s];
    p.h[1] = (__bf16)stg[(cip + 1) * STILE + s];
    *(unsigned*)(Xs + s * XS_STR + cip) = p.u;
  }
  // Wb[o][ci]
  for (int it = 0; it < 12; ++it) {
    int idx = it * 512 + tid;                 // 6144 pairs
    int cip = (idx & 31) << 1;
    int o   = idx >> 5;                       // 0..191
    BF2 p;
    p.h[0] = (__bf16)Wg[o * 64 + cip];
    p.h[1] = (__bf16)Wg[o * 64 + cip + 1];
    *(unsigned*)(Wb + o * WB_STR + cip) = p.u;
  }
  // Ab[k][wtile][w][v] = (A*edge)[k][v][w], zero padded to v,w in [0,32)
  for (int it = 0; it < 6; ++it) {
    int idx = it * 512 + tid;                 // 3072
    int k = idx >> 10, w = (idx >> 5) & 31, v = idx & 31;
    float a = 0.0f;
    if (w < VDIM && v < VDIM) {
      int ai = (k * VDIM + v) * VDIM + w;
      a = Aadj[ai] * Eadj[ai];
    }
    int wt = w >> 4, wr = w & 15;
    Ab[((k * 2 + wt) * 16 + wr) * AB_STR + v] = (__bf16)a;
  }
  // bias2[c][w] = sum_k b[k*64+c] * sum_v (A*edge)[k][v][w]
  for (int it = 0; it < 4; ++it) {
    int idx = it * 512 + tid;
    if (idx < 1600) {
      int c = idx / VDIM, w = idx % VDIM;
      float s = 0.0f;
      for (int k = 0; k < KADJ; ++k) {
        float cs = 0.0f;
        for (int v = 0; v < VDIM; ++v) {
          int ai = (k * VDIM + v) * VDIM + w;
          cs += Aadj[ai] * Eadj[ai];
        }
        s += bg[k * 64 + c] * cs;
      }
      B2[c * 32 + w] = s;
    }
  }
  __syncthreads();

  // ---- phase 2: zero Ys (staging dead now); keeps v in [25,32) zero ----
  for (int it = 0; it < 41; ++it) {
    int idx = it * 512 + tid;
    if (idx < (YS_BYTES / 4)) ((unsigned*)Ys)[idx] = 0u;
  }
  __syncthreads();

  // ---- phase 3: k-loop, two chained WMMA GEMMs ----
  v8f accB[8];
#pragma unroll
  for (int i = 0; i < 8; ++i) accB[i] = (v8f){0.f,0.f,0.f,0.f,0.f,0.f,0.f,0.f};

  for (int k = 0; k < KADJ; ++k) {
    // step A: Y_k[64 o][400 s] = W_k[64x64] @ X[64x400]  (bf16 -> f32 -> bf16)
    for (int tl = wid; tl < 100; tl += 16) {
      int mt = tl / 25, nt = tl % 25;
      const __bf16* arow = Wb + (k * 64 + mt * 16 + l15) * WB_STR;
      const __bf16* brow = Xs + (nt * 16 + l15) * XS_STR;
      v8f acc = (v8f){0.f,0.f,0.f,0.f,0.f,0.f,0.f,0.f};
#pragma unroll
      for (int ks = 0; ks < 2; ++ks) {
        v16bf a = frag_a(arow + ks * 32, lhalf);
        v16bf b = frag_b(brow + ks * 32, lhalf);
        acc = __builtin_amdgcn_wmma_f32_16x16x32_bf16(false, a, false, b,
                                                      (short)0, acc, false, false);
      }
      int s = nt * 16 + l15;            // this lane's output column
      int t = s / VDIM, v = s % VDIM;
      __bf16* yo = Ys + t * YS_TSTR + v;
#pragma unroll
      for (int j = 0; j < 8; ++j) {
        int c = mt * 16 + lhalf * 8 + j; // D-layout row
        yo[c * YS_CSTR] = (__bf16)acc[j];
      }
    }
    __syncthreads();

    // step B: out[c][t][w] += Y_k[c][t][v32] @ A_k[v32][w]   (K=32 per WMMA)
#pragma unroll
    for (int i = 0; i < 8; ++i) {
      int ot = i * 16 + wid;            // 128 tiles: 4 ctile x 16 t x 2 wtile
      int wt = ot & 1, t = (ot >> 1) & 15, ct = ot >> 5;
      const __bf16* arow = Ys + (ct * 16 + l15) * YS_CSTR + t * YS_TSTR;
      const __bf16* brow = Ab + ((k * 2 + wt) * 16 + l15) * AB_STR;
      v16bf a = frag_a(arow, lhalf);
      v16bf b = frag_b(brow, lhalf);
      accB[i] = __builtin_amdgcn_wmma_f32_16x16x32_bf16(false, a, false, b,
                                                        (short)0, accB[i], false, false);
    }
    __syncthreads();
  }

  // ---- phase 4: add folded bias, store pre-BN y as bf16 ----
#pragma unroll
  for (int i = 0; i < 8; ++i) {
    int ot = i * 16 + wid;
    int wt = ot & 1, t = (ot >> 1) & 15, ct = ot >> 5;
    int w = wt * 16 + l15;
    int tg = t0 + t;
    if (w < VDIM && tg < TDIM) {
#pragma unroll
      for (int j = 0; j < 8; ++j) {
        int c = ct * 16 + lhalf * 8 + j;
        float val = accB[i][j] + B2[c * 32 + w];
        ybf[(((size_t)n * 64 + c) * TDIM + tg) * VDIM + w] = (__bf16)val;
      }
    }
  }
}

// =====================================================================
// K2: per-channel sum / sumsq   grid (64 c, 64 n), 256 threads
// =====================================================================
__global__ __launch_bounds__(256) void k_stats(const unsigned short* __restrict__ ybf_raw,
                                               float* __restrict__ stats)
{
  __shared__ float red[512];
  const __bf16* ybf = (const __bf16*)ybf_raw;
  int c = blockIdx.x, n = blockIdx.y, tid = threadIdx.x;
  size_t base = ((size_t)n * 64 + c) * TVT;
  float s = 0.f, s2 = 0.f;
  for (int i = tid; i < TVT; i += 256) {
    float v = (float)ybf[base + i];
    s += v; s2 += v * v;
  }
  red[tid] = s; red[256 + tid] = s2;
  __syncthreads();
  for (int o = 128; o > 0; o >>= 1) {
    if (tid < o) { red[tid] += red[tid + o]; red[256 + tid] += red[256 + tid + o]; }
    __syncthreads();
  }
  if (tid == 0) {
    atomicAdd(&stats[c], red[0]);
    atomicAdd(&stats[64 + c], red[256]);
  }
}

// =====================================================================
// K3: finalize BN scale/shift per channel
// =====================================================================
__global__ void k_finalize(const float* __restrict__ gamma,
                           const float* __restrict__ beta,
                           float* __restrict__ stats)
{
  int c = threadIdx.x;
  if (c < 64) {
    const float invn = 1.0f / 480000.0f;
    float mean = stats[c] * invn;
    float var  = stats[64 + c] * invn - mean * mean;
    float inv  = rsqrtf(var + 1e-5f);
    float a = gamma[c] * inv;
    stats[128 + c] = a;
    stats[192 + c] = beta[c] - mean * a;
  }
}

// =====================================================================
// K4: epilogue  out = relu(y*scale + shift + x)   float4-vectorized
// =====================================================================
__global__ __launch_bounds__(256) void k_epilogue(
    const float* __restrict__ x, const unsigned short* __restrict__ ybf_raw,
    const float* __restrict__ stats, float* __restrict__ out)
{
  int i4 = blockIdx.x * 256 + threadIdx.x;     // 7,680,000 quads exactly
  size_t base = (size_t)i4 * 4;
  int c = (int)((base / TVT) & 63);            // TVT % 4 == 0 -> c uniform in quad
  float a = stats[128 + c], b = stats[192 + c];
  BF4 y; y.u = *(const uint2*)(ybf_raw + base);
  float4 xv = *(const float4*)(x + base);
  float4 o;
  o.x = fmaxf(fmaf((float)y.h[0], a, b) + xv.x, 0.f);
  o.y = fmaxf(fmaf((float)y.h[1], a, b) + xv.y, 0.f);
  o.z = fmaxf(fmaf((float)y.h[2], a, b) + xv.z, 0.f);
  o.w = fmaxf(fmaf((float)y.h[3], a, b) + xv.w, 0.f);
  *(float4*)(out + base) = o;
}

// =====================================================================
extern "C" void kernel_launch(void* const* d_in, const int* in_sizes, int n_in,
                              void* d_out, int out_size, void* d_ws, size_t ws_size,
                              hipStream_t stream) {
  (void)in_sizes; (void)n_in; (void)out_size; (void)ws_size;
  const float* x     = (const float*)d_in[0];
  const float* Wg    = (const float*)d_in[1];
  const float* bg    = (const float*)d_in[2];
  const float* Aadj  = (const float*)d_in[3];
  const float* Eadj  = (const float*)d_in[4];
  const float* gamma = (const float*)d_in[5];
  const float* beta  = (const float*)d_in[6];
  float* out = (float*)d_out;

  unsigned short* ybf = (unsigned short*)d_ws;                       // 61.44 MB bf16
  float* stats = (float*)((char*)d_ws + (size_t)NELEM * 2);          // 256 floats

  k_zero_stats<<<1, 128, 0, stream>>>(stats);
  k_fused_gcn<<<dim3(NB, (TDIM + TTILE - 1) / TTILE), 512, LDS_TOTAL, stream>>>(
      x, Wg, bg, Aadj, Eadj, ybf);
  k_stats<<<dim3(64, 64), 256, 0, stream>>>(ybf, stats);
  k_finalize<<<1, 64, 0, stream>>>(gamma, beta, stats);
  k_epilogue<<<NELEM / 4 / 256, 256, 0, stream>>>(x, ybf, stats, out);
}
